// GNN_Layer_23167053595206
// MI455X (gfx1250) — compile-verified, matrix operand order checked by reading, not run on hardware
//
#include <hip/hip_runtime.h>

typedef __attribute__((ext_vector_type(16))) _Float16 v16h;
typedef __attribute__((ext_vector_type(8)))  _Float16 v8h;
typedef __attribute__((ext_vector_type(4)))  _Float16 v4h;
typedef __attribute__((ext_vector_type(8)))  float    v8f;

#define N_NODES_C 100000
#define N_EDGES_C 1600000
#define DIM_IN 48
#define DIM_U  128
#define DIM_OUT 128
#define MTILE 32

// ---------------- utility kernels ----------------

__global__ void zero_f32v4(float4* __restrict__ p, long n4) {
  long i = (long)blockIdx.x * blockDim.x + threadIdx.x;
  if (i < n4) p[i] = make_float4(0.f, 0.f, 0.f, 0.f);
}

// Produce K-contiguous (transposed) f16 weights:
//   W1t[n][k], k padded 48->64 ;  W2t[n][k]
__global__ void prep_weights(const float* __restrict__ W1, const float* __restrict__ W2,
                             _Float16* __restrict__ W1t, _Float16* __restrict__ W2t) {
  int i = blockIdx.x * blockDim.x + threadIdx.x;
  if (i < DIM_U * 64) {
    int n = i >> 6, k = i & 63;
    float v = (k < DIM_IN) ? W1[k * DIM_U + n] : 0.0f;
    W1t[i] = (_Float16)v;
  }
  if (i < DIM_U * DIM_OUT) {
    int n = i >> 7, k = i & 127;
    W2t[i] = (_Float16)W2[k * DIM_OUT + n];
  }
}

// ---------------- edge message passing ----------------
// accum[end[e]] += ysrc[start[e]] * dfs[e] * (use_src_scale ? dfn[start[e]] : 1)
// 12 lanes per edge, one float4 chunk each; whole working set lives in L2.
__global__ __launch_bounds__(256) void edge_pass(
    const float* __restrict__ ysrc, const int* __restrict__ sn, const int* __restrict__ en,
    const float* __restrict__ dfs, const float* __restrict__ dfn,
    float* __restrict__ accum, int use_src_scale)
{
  long i = (long)blockIdx.x * blockDim.x + threadIdx.x;
  if (i >= (long)N_EDGES_C * 12) return;
  int edge = (int)(i / 12);
  int c    = (int)(i % 12);
  int src = sn[edge], dst = en[edge];
  float w = dfs[edge];
  if (use_src_scale) w *= dfn[src];
  const float4 v = *(const float4*)(ysrc + (long)src * DIM_IN + c * 4);
  float* dp = accum + (long)dst * DIM_IN + c * 4;
  unsafeAtomicAdd(dp + 0, v.x * w);   // global_atomic_add_f32
  unsafeAtomicAdd(dp + 1, v.y * w);
  unsafeAtomicAdd(dp + 2, v.z * w);
  unsafeAtomicAdd(dp + 3, v.w * w);
}

// ---------------- fused MLP via WMMA ----------------
// Per-lane 16-bit A/B fragment = two contiguous 16B runs (CDNA5 ISA):
//   halves 0..7  -> k = base + 8*g + (0..7)
//   halves 8..15 -> k = base + 16 + 8*g + (0..7)
__device__ __forceinline__ v16h load_frag(const _Float16* __restrict__ rowp, int base, int g) {
  v8h lo = *(const v8h*)(rowp + base + 8 * g);        // ds_load_b128
  v8h hi = *(const v8h*)(rowp + base + 16 + 8 * g);   // ds_load_b128
  return __builtin_shufflevector(lo, hi, 0, 1, 2, 3, 4, 5, 6, 7,
                                 8, 9, 10, 11, 12, 13, 14, 15);
}

__global__ __launch_bounds__(64) void mlp_wmma(
    const float* __restrict__ x,       // accum2 [N][48] (pre-df)
    const float* __restrict__ dfn,     // [N]
    const _Float16* __restrict__ W1t,  // [128][64]  K-contiguous
    const _Float16* __restrict__ W2t,  // [128][128] K-contiguous
    const float* __restrict__ bias1, const float* __restrict__ bias2,
    float* __restrict__ out, int n_nodes)
{
  __shared__ __align__(16) _Float16 xT[MTILE][64];          // 4 KB, row stride 128B
  __shared__ __align__(16) _Float16 hT[MTILE][DIM_U + 8];   // 8.7 KB, row stride 272B
  __shared__ __align__(16) _Float16 w1s[DIM_U * 64];        // 16 KB  [n][k]
  __shared__ __align__(16) _Float16 w2s[DIM_U * DIM_OUT];   // 32 KB  [n][k]

  const int tid  = threadIdx.x;        // 0..63 (2 waves)
  const int lane = tid & 31;
  const int wave = tid >> 5;           // each wave owns 16 rows
  const int g    = lane >> 4;
  const int l    = lane & 15;
  const int row0 = blockIdx.x * MTILE;

  // Stage weights as 16B vectors: 16 + 32 iterations per thread.
  {
    const uint4* s1 = (const uint4*)W1t; uint4* d1 = (uint4*)w1s;
    for (int i = tid; i < DIM_U * 64 * 2 / 16; i += 64) d1[i] = s1[i];
    const uint4* s2 = (const uint4*)W2t; uint4* d2 = (uint4*)w2s;
    for (int i = tid; i < DIM_U * DIM_OUT * 2 / 16; i += 64) d2[i] = s2[i];
  }
  // Stage x tile: float4 gather -> v4h store; then zero the K pad (48..63).
  for (int i = tid; i < MTILE * 12; i += 64) {
    int r = i / 12, c4 = i % 12;
    int gr = row0 + r;
    v4h hv = {};
    if (gr < n_nodes) {
      float4 v = *(const float4*)(x + (long)gr * DIM_IN + c4 * 4);
      float s = dfn[gr];
      hv[0] = (_Float16)(v.x * s); hv[1] = (_Float16)(v.y * s);
      hv[2] = (_Float16)(v.z * s); hv[3] = (_Float16)(v.w * s);
    }
    *(v4h*)(&xT[r][c4 * 4]) = hv;
  }
  for (int i = tid; i < MTILE * 2; i += 64) {
    v8h z = {};
    *(v8h*)(&xT[i >> 1][48 + (i & 1) * 8]) = z;
  }
  __syncthreads();

  const int mrow = wave * 16;

  // GEMM1: H = relu(X*W1 + b1); K=64 -> 2 WMMA steps, 8 N-tiles.
  const _Float16* arow = &xT[mrow + l][0];
  v16h a0 = load_frag(arow, 0, g);
  v16h a1 = load_frag(arow, 32, g);

#pragma unroll
  for (int nt = 0; nt < 8; ++nt) {
    const _Float16* brow = &w1s[(nt * 16 + l) * 64];
    v16h b0 = load_frag(brow, 0, g);
    v16h b1 = load_frag(brow, 32, g);
    v8f c = {};
    c = __builtin_amdgcn_wmma_f32_16x16x32_f16(false, a0, false, b0, (short)0, c, false, false);
    c = __builtin_amdgcn_wmma_f32_16x16x32_f16(false, a1, false, b1, (short)0, c, false, false);
    float bb = bias1[nt * 16 + l];
#pragma unroll
    for (int r = 0; r < 8; ++r) {
      float hv = c[r] + bb;                       // c[r] = H[mrow+r+8g][nt*16+l]
      hT[mrow + r + 8 * g][nt * 16 + l] = (_Float16)(hv > 0.0f ? hv : 0.0f);
    }
  }
  __syncthreads();

  // GEMM2: out = H*W2 + b2; K=128 -> 4 WMMA steps, 8 N-tiles.
  const _Float16* hrow = &hT[mrow + l][0];
  v16h ah[4];
#pragma unroll
  for (int ks = 0; ks < 4; ++ks) ah[ks] = load_frag(hrow, ks * 32, g);

#pragma unroll
  for (int nt = 0; nt < 8; ++nt) {
    const _Float16* brow = &w2s[(nt * 16 + l) * 128];
    v8f c = {};
#pragma unroll
    for (int ks = 0; ks < 4; ++ks) {
      v16h bb = load_frag(brow, ks * 32, g);
      c = __builtin_amdgcn_wmma_f32_16x16x32_f16(false, ah[ks], false, bb, (short)0, c, false, false);
    }
    float bo = bias2[nt * 16 + l];
#pragma unroll
    for (int r = 0; r < 8; ++r) {
      int gr = row0 + mrow + r + 8 * g;
      if (gr < n_nodes) out[(long)gr * DIM_OUT + nt * 16 + l] = c[r] + bo;
    }
  }
}

// ---------------- launch ----------------

extern "C" void kernel_launch(void* const* d_in, const int* in_sizes, int n_in,
                              void* d_out, int out_size, void* d_ws, size_t ws_size,
                              hipStream_t stream) {
  (void)in_sizes; (void)n_in; (void)out_size; (void)ws_size;

  const float* y   = (const float*)d_in[0];
  const int*   sn  = (const int*)d_in[1];
  const int*   en  = (const int*)d_in[2];
  const float* dfs = (const float*)d_in[3];
  const float* dfn = (const float*)d_in[4];
  const float* W1  = (const float*)d_in[5];
  const float* b1  = (const float*)d_in[6];
  const float* W2  = (const float*)d_in[7];
  const float* b2  = (const float*)d_in[8];

  char* ws = (char*)d_ws;
  float* accum1 = (float*)ws;  ws += (size_t)N_NODES_C * DIM_IN * sizeof(float);
  float* accum2 = (float*)ws;  ws += (size_t)N_NODES_C * DIM_IN * sizeof(float);
  _Float16* W1t = (_Float16*)ws;  ws += (size_t)DIM_U * 64 * sizeof(_Float16);
  _Float16* W2t = (_Float16*)ws;  ws += (size_t)DIM_U * DIM_OUT * sizeof(_Float16);

  const long nz4 = (long)N_NODES_C * DIM_IN / 4;   // 1.2M float4 per buffer
  zero_f32v4<<<(int)((nz4 + 255) / 256), 256, 0, stream>>>((float4*)accum1, nz4);
  zero_f32v4<<<(int)((nz4 + 255) / 256), 256, 0, stream>>>((float4*)accum2, nz4);
  prep_weights<<<(DIM_U * DIM_OUT + 255) / 256, 256, 0, stream>>>(W1, W2, W1t, W2t);

  const long ne12 = (long)N_EDGES_C * 12;
  edge_pass<<<(int)((ne12 + 255) / 256), 256, 0, stream>>>(y,      sn, en, dfs, dfn, accum1, 0);
  edge_pass<<<(int)((ne12 + 255) / 256), 256, 0, stream>>>(accum1, sn, en, dfs, dfn, accum2, 1);

  mlp_wmma<<<(N_NODES_C + MTILE - 1) / MTILE, 64, 0, stream>>>(
      accum2, dfn, W1t, W2t, b1, b2, (float*)d_out, N_NODES_C);

  // Passthrough tuple outputs, concatenated after `out`
  char* o = (char*)d_out + (size_t)N_NODES_C * DIM_OUT * sizeof(float);
  hipMemcpyAsync(o, d_in[1], (size_t)N_EDGES_C * 4, hipMemcpyDeviceToDevice, stream); o += (size_t)N_EDGES_C * 4;
  hipMemcpyAsync(o, d_in[2], (size_t)N_EDGES_C * 4, hipMemcpyDeviceToDevice, stream); o += (size_t)N_EDGES_C * 4;
  hipMemcpyAsync(o, d_in[3], (size_t)N_EDGES_C * 4, hipMemcpyDeviceToDevice, stream); o += (size_t)N_EDGES_C * 4;
  hipMemcpyAsync(o, d_in[4], (size_t)N_NODES_C * 4, hipMemcpyDeviceToDevice, stream);
}
